// Att_23313082483285
// MI455X (gfx1250) — matrix operand order repaired and checked by reading, not compiled, not run on hardware
//
#include <hip/hip_runtime.h>

typedef __attribute__((ext_vector_type(16))) _Float16 v16h;
typedef __attribute__((ext_vector_type(8)))  _Float16 v8h;
typedef __attribute__((ext_vector_type(8)))  float    v8f;

union HV16 { v16h v; v8h h[2]; _Float16 e[16]; };
union HPK4 { _Float16 h[4]; uint2 u; };

#define ASTR 392      // halves per A-tile row in LDS (384 + 8 pad; row stride 784B = 49*16B)
#define MSTR 136      // halves per row for 128-wide tiles (272B = 17*16B)

// fragment-layout weight offsets in d_ws (in halves)
#define FR_DISTW2 0
#define FR_QW     16384
#define FR_CTXW1  32768
#define FR_CTXW2  81920
#define FR_AGTW   98304
#define FR_LINW   114688
#define FR_TOTAL  131072

__device__ __forceinline__ void zacc(v8f a[8]) {
#pragma unroll
  for (int nt = 0; nt < 8; ++nt)
#pragma unroll
    for (int r = 0; r < 8; ++r) a[nt][r] = 0.0f;
}

__device__ __forceinline__ v8f do_wmma(const HV16& a, const HV16& b, v8f c) {
  return __builtin_amdgcn_wmma_f32_16x16x32_f16(false, a.v, false, b.v, (short)0, c,
                                                false, false);
}

// C(32x128) += A(32 x 32*nK f16 in LDS) x B(frag-layout f16 in global).
// Two 16-row A sub-tiles share each B fragment; B is prefetched one step ahead.
__device__ __forceinline__ void mm2(const _Float16* aLds, int stride, int colBase, int nK,
                                    const _Float16* __restrict__ bFrag,
                                    v8f acc0[8], v8f acc1[8], int lane) {
  const int row = lane & 15, kg = lane >> 4;
  for (int kc = 0; kc < nK; ++kc) {
    HV16 a0, a1;
    const _Float16* ap0 = aLds + row * stride + colBase + kc * 32 + kg * 8;
    const _Float16* ap1 = ap0 + 16 * stride;
    a0.h[0] = *(const v8h*)(ap0); a0.h[1] = *(const v8h*)(ap0 + 16);
    a1.h[0] = *(const v8h*)(ap1); a1.h[1] = *(const v8h*)(ap1 + 16);
    const _Float16* bp0 = bFrag + (((kc * 8) * 32 + lane) << 4);
    HV16 bc;
    bc.h[0] = *(const v8h*)(bp0); bc.h[1] = *(const v8h*)(bp0 + 8);
#pragma unroll
    for (int nt = 0; nt < 8; ++nt) {
      HV16 bn;
      if (nt < 7) {                 // prefetch next fragment into a distinct buffer
        const _Float16* bq = bFrag + (((kc * 8 + nt + 1) * 32 + lane) << 4);
        bn.h[0] = *(const v8h*)(bq); bn.h[1] = *(const v8h*)(bq + 8);
      }
      acc0[nt] = do_wmma(a0, bc, acc0[nt]);
      acc1[nt] = do_wmma(a1, bc, acc1[nt]);
      if (nt < 7) bc = bn;
    }
  }
}

// GroupNorm(ng=1) over the 128 channels of each of 16 rows held in C fragments.
__device__ __forceinline__ void groupnorm(v8f acc[8], const float* __restrict__ gw,
                                          const float* __restrict__ gb, int lane, bool relu) {
  const int ln = lane & 15;
  float gwv[8], gbv[8];
#pragma unroll
  for (int nt = 0; nt < 8; ++nt) { gwv[nt] = gw[nt * 16 + ln]; gbv[nt] = gb[nt * 16 + ln]; }
#pragma unroll
  for (int r = 0; r < 8; ++r) {
    float s = 0.0f, ss = 0.0f;
#pragma unroll
    for (int nt = 0; nt < 8; ++nt) { float x = acc[nt][r]; s += x; ss += x * x; }
#pragma unroll
    for (int m = 1; m < 16; m <<= 1) {       // reduce across the 16-lane half
      s  += __shfl_xor(s,  m, 32);
      ss += __shfl_xor(ss, m, 32);
    }
    float mean = s * 0.0078125f;
    float var  = ss * 0.0078125f - mean * mean;
    float inv  = rsqrtf(var + 1e-5f);
#pragma unroll
    for (int nt = 0; nt < 8; ++nt) {
      float y = (acc[nt][r] - mean) * inv * gwv[nt] + gbv[nt];
      if (relu) y = fmaxf(y, 0.0f);
      acc[nt][r] = y;
    }
  }
}

// write one 16-row C-fragment set back to an f16 row-major LDS A-tile
__device__ __forceinline__ void store_c_to_lds(const v8f acc[8], _Float16* aLds, int stride,
                                               int rowOff, int colBase, int lane) {
  const int ln = lane & 15, kg = lane >> 4;
#pragma unroll
  for (int r = 0; r < 8; ++r) {
    int row = rowOff + r + 8 * kg;
#pragma unroll
    for (int nt = 0; nt < 8; ++nt)
      aLds[row * stride + colBase + nt * 16 + ln] = (_Float16)acc[nt][r];
  }
}

// ---------------- kernel 1: weights -> f16 B-fragment layout ----------------
__global__ __launch_bounds__(256) void build_frags(
    const float* __restrict__ dist_w2, const float* __restrict__ q_w,
    const float* __restrict__ ctx_w1,  const float* __restrict__ ctx_w2,
    const float* __restrict__ agt_w,   const float* __restrict__ lin_w,
    _Float16* __restrict__ frags) {
  int idx = blockIdx.x * 256 + threadIdx.x;
  if (idx >= FR_TOTAL) return;
  const float* src; int local, base;
  if      (idx < 16384)  { src = dist_w2; local = idx;          base = FR_DISTW2; }
  else if (idx < 32768)  { src = q_w;     local = idx - 16384;  base = FR_QW; }
  else if (idx < 81920)  { src = ctx_w1;  local = idx - 32768;  base = FR_CTXW1; }
  else if (idx < 98304)  { src = ctx_w2;  local = idx - 81920;  base = FR_CTXW2; }
  else if (idx < 114688) { src = agt_w;   local = idx - 98304;  base = FR_AGTW; }
  else                   { src = lin_w;   local = idx - 114688; base = FR_LINW; }
  int k = local >> 7, n = local & 127;
  int kc = k >> 5, kk = k & 31, nt = n >> 4, nn = n & 15;
  int lane = nn + (((kk >> 3) & 1) << 4);
  int hh = (kk & 7) + ((kk >> 4) << 3);
  frags[base + (((kc * 8 + nt) * 32 + lane) << 4) + hh] = (_Float16)src[local];
}

// ---------------- kernel 2: acc = agts @ agt_w (32 rows per wave) ----------------
__global__ __launch_bounds__(64) void init_acc_kernel(
    const float* __restrict__ agts, const _Float16* __restrict__ frags,
    float* __restrict__ accBuf, int nAgt) {
  int lane = threadIdx.x & 31, wid = threadIdx.x >> 5;
  int tile = blockIdx.x * 2 + wid;
  if (tile * 32 >= nAgt) return;
  int row0 = tile * 32, myRow = lane & 15, kg = lane >> 4, ln = lane & 15;
  v8f acc0[8], acc1[8]; zacc(acc0); zacc(acc1);
  for (int kc = 0; kc < 4; ++kc) {
    HV16 a0, a1;
    const float* s0 = agts + (size_t)(row0 + myRow) * 128 + kc * 32 + kg * 8;
    const float* s1 = s0 + 16 * 128;
#pragma unroll
    for (int i = 0; i < 8; ++i) { a0.e[i] = (_Float16)s0[i];      a1.e[i] = (_Float16)s1[i]; }
#pragma unroll
    for (int i = 0; i < 8; ++i) { a0.e[8 + i] = (_Float16)s0[16 + i];
                                  a1.e[8 + i] = (_Float16)s1[16 + i]; }
    const _Float16* bp0 = frags + FR_AGTW + (((kc * 8) * 32 + lane) << 4);
    HV16 bc; bc.h[0] = *(const v8h*)(bp0); bc.h[1] = *(const v8h*)(bp0 + 8);
#pragma unroll
    for (int nt = 0; nt < 8; ++nt) {
      HV16 bn;
      if (nt < 7) {
        const _Float16* bq = frags + FR_AGTW + (((kc * 8 + nt + 1) * 32 + lane) << 4);
        bn.h[0] = *(const v8h*)(bq); bn.h[1] = *(const v8h*)(bq + 8);
      }
      acc0[nt] = do_wmma(a0, bc, acc0[nt]);
      acc1[nt] = do_wmma(a1, bc, acc1[nt]);
      if (nt < 7) bc = bn;
    }
  }
#pragma unroll
  for (int r = 0; r < 8; ++r)
#pragma unroll
    for (int nt = 0; nt < 8; ++nt) {
      accBuf[(size_t)(row0 + r + 8 * kg) * 128 + nt * 16 + ln]      = acc0[nt][r];
      accBuf[(size_t)(row0 + 16 + r + 8 * kg) * 128 + nt * 16 + ln] = acc1[nt][r];
    }
}

// ---------------- kernel 3: per-edge message + scatter-add (32 edges/wave) ----------------
__global__ __launch_bounds__(64) void edge_kernel(
    const float* __restrict__ agts, const float* __restrict__ ctx,
    const float* __restrict__ agt_ctrs, const float* __restrict__ ctx_ctrs,
    const int* __restrict__ hi, const int* __restrict__ wi,
    const float* __restrict__ dist_w1, const float* __restrict__ dist_b1,
    const float* __restrict__ dist_gw, const float* __restrict__ dist_gb,
    const float* __restrict__ q_gw, const float* __restrict__ q_gb,
    const float* __restrict__ ctx_gw, const float* __restrict__ ctx_gb,
    const _Float16* __restrict__ frags, float* __restrict__ accBuf, int E) {
  __shared__ _Float16 smem[2 * 32 * ASTR];   // 50176 B, one 32-row A-tile per wave
  int lane = threadIdx.x & 31, wid = threadIdx.x >> 5;
  int tile = blockIdx.x * 2 + wid;
  if (tile * 32 >= E) return;                // wave-uniform
  _Float16* aLds = smem + wid * (32 * ASTR);
  int e0 = tile * 32;
  int myRow = lane & 15, kg = lane >> 4, ln = lane & 15;

  int hrow[2], wrow[2];
  float rx[2], ry[2];
#pragma unroll
  for (int s = 0; s < 2; ++s) {
    int ee = e0 + s * 16 + myRow; if (ee > E - 1) ee = E - 1;
    hrow[s] = hi[ee]; wrow[s] = wi[ee];
    float2 acv = ((const float2*)agt_ctrs)[hrow[s]];
    float2 ccv = ((const float2*)ctx_ctrs)[wrow[s]];
    rx[s] = acv.x - ccv.x; ry[s] = acv.y - ccv.y;
  }

  // phase 1: d1 = relu(rel @ dist_w1 + b1) -> A cols [0,128)
  {
    int c0 = kg * 64;
    for (int j = 0; j < 64; j += 4) {
      float4 w0 = *(const float4*)(dist_w1 + c0 + j);
      float4 w1 = *(const float4*)(dist_w1 + 128 + c0 + j);
      float4 bb = *(const float4*)(dist_b1 + c0 + j);
#pragma unroll
      for (int s = 0; s < 2; ++s) {
        HPK4 p;
        p.h[0] = (_Float16)fmaxf(rx[s] * w0.x + ry[s] * w1.x + bb.x, 0.0f);
        p.h[1] = (_Float16)fmaxf(rx[s] * w0.y + ry[s] * w1.y + bb.y, 0.0f);
        p.h[2] = (_Float16)fmaxf(rx[s] * w0.z + ry[s] * w1.z + bb.z, 0.0f);
        p.h[3] = (_Float16)fmaxf(rx[s] * w0.w + ry[s] * w1.w + bb.w, 0.0f);
        *(uint2*)(aLds + (s * 16 + myRow) * ASTR + c0 + j) = p.u;
      }
    }
  }
  __builtin_amdgcn_sched_barrier(0);
  v8f acc0[8], acc1[8];

  // M1: d = relu(gn(d1 @ dist_w2)) -> back into A cols [0,128)
  zacc(acc0); zacc(acc1);
  mm2(aLds, ASTR, 0, 4, frags + FR_DISTW2, acc0, acc1, lane);
  groupnorm(acc0, dist_gw, dist_gb, lane, true);
  groupnorm(acc1, dist_gw, dist_gb, lane, true);
  __builtin_amdgcn_sched_barrier(0);
  store_c_to_lds(acc0, aLds, ASTR, 0, 0, lane);
  store_c_to_lds(acc1, aLds, ASTR, 16, 0, lane);
  __builtin_amdgcn_sched_barrier(0);

  // phase 2: gather agts[hi] -> A cols [128,256)
  {
    int c0 = kg * 64;
#pragma unroll
    for (int s = 0; s < 2; ++s) {
      const float* src = agts + (size_t)hrow[s] * 128 + c0;
      for (int j = 0; j < 64; j += 4) {
        float4 x = *(const float4*)(src + j);
        HPK4 p;
        p.h[0] = (_Float16)x.x; p.h[1] = (_Float16)x.y;
        p.h[2] = (_Float16)x.z; p.h[3] = (_Float16)x.w;
        *(uint2*)(aLds + (s * 16 + myRow) * ASTR + 128 + c0 + j) = p.u;
      }
    }
  }
  __builtin_amdgcn_sched_barrier(0);

  // M2: q = relu(gn(agts[hi] @ q_w)) -> back into A cols [128,256)
  zacc(acc0); zacc(acc1);
  mm2(aLds, ASTR, 128, 4, frags + FR_QW, acc0, acc1, lane);
  groupnorm(acc0, q_gw, q_gb, lane, true);
  groupnorm(acc1, q_gw, q_gb, lane, true);
  __builtin_amdgcn_sched_barrier(0);
  store_c_to_lds(acc0, aLds, ASTR, 0, 128, lane);
  store_c_to_lds(acc1, aLds, ASTR, 16, 128, lane);
  __builtin_amdgcn_sched_barrier(0);

  // phase 3: gather ctx[wi] -> A cols [256,384)
  {
    int c0 = kg * 64;
#pragma unroll
    for (int s = 0; s < 2; ++s) {
      const float* src = ctx + (size_t)wrow[s] * 128 + c0;
      for (int j = 0; j < 64; j += 4) {
        float4 x = *(const float4*)(src + j);
        HPK4 p;
        p.h[0] = (_Float16)x.x; p.h[1] = (_Float16)x.y;
        p.h[2] = (_Float16)x.z; p.h[3] = (_Float16)x.w;
        *(uint2*)(aLds + (s * 16 + myRow) * ASTR + 256 + c0 + j) = p.u;
      }
    }
  }
  __builtin_amdgcn_sched_barrier(0);

  // M3: m = relu(gn(concat @ ctx_w1)) -> into A cols [0,128)
  zacc(acc0); zacc(acc1);
  mm2(aLds, ASTR, 0, 12, frags + FR_CTXW1, acc0, acc1, lane);
  groupnorm(acc0, ctx_gw, ctx_gb, lane, true);
  groupnorm(acc1, ctx_gw, ctx_gb, lane, true);
  __builtin_amdgcn_sched_barrier(0);
  store_c_to_lds(acc0, aLds, ASTR, 0, 0, lane);
  store_c_to_lds(acc1, aLds, ASTR, 16, 0, lane);
  __builtin_amdgcn_sched_barrier(0);

  // M4: m2 = m @ ctx_w2, then scatter-add to acc[hi]
  zacc(acc0); zacc(acc1);
  mm2(aLds, ASTR, 0, 4, frags + FR_CTXW2, acc0, acc1, lane);
#pragma unroll
  for (int s = 0; s < 2; ++s) {
    v8f* acc = s ? acc1 : acc0;
#pragma unroll
    for (int r = 0; r < 8; ++r) {
      int m = r + 8 * kg;
      int agent = __shfl(hrow[s], m, 32);
      if (e0 + s * 16 + m < E) {
        float* dst = accBuf + (size_t)agent * 128 + ln;
#pragma unroll
        for (int nt = 0; nt < 8; ++nt) atomicAdd(dst + nt * 16, acc[nt][r]);
      }
    }
  }
}

// ---------------- kernel 4: out = relu(gn(gn_relu(acc) @ lin_w) + agts) ----------------
__global__ __launch_bounds__(64) void final_kernel(
    const float* __restrict__ agts, const _Float16* __restrict__ frags,
    const float* __restrict__ norm_w, const float* __restrict__ norm_b,
    const float* __restrict__ lin_gw, const float* __restrict__ lin_gb,
    float* __restrict__ accBuf, int nAgt) {
  __shared__ _Float16 smem[2 * 32 * MSTR];
  int lane = threadIdx.x & 31, wid = threadIdx.x >> 5;
  int tile = blockIdx.x * 2 + wid;
  if (tile * 32 >= nAgt) return;
  _Float16* aLds = smem + wid * (32 * MSTR);
  int row0 = tile * 32, kg = lane >> 4, ln = lane & 15;

  v8f acc0[8], acc1[8];
#pragma unroll
  for (int r = 0; r < 8; ++r)
#pragma unroll
    for (int nt = 0; nt < 8; ++nt) {
      acc0[nt][r] = accBuf[(size_t)(row0 + r + 8 * kg) * 128 + nt * 16 + ln];
      acc1[nt][r] = accBuf[(size_t)(row0 + 16 + r + 8 * kg) * 128 + nt * 16 + ln];
    }

  groupnorm(acc0, norm_w, norm_b, lane, true);
  groupnorm(acc1, norm_w, norm_b, lane, true);
  __builtin_amdgcn_sched_barrier(0);
  store_c_to_lds(acc0, aLds, MSTR, 0, 0, lane);
  store_c_to_lds(acc1, aLds, MSTR, 16, 0, lane);
  __builtin_amdgcn_sched_barrier(0);

  v8f o0[8], o1[8]; zacc(o0); zacc(o1);
  mm2(aLds, MSTR, 0, 4, frags + FR_LINW, o0, o1, lane);
  groupnorm(o0, lin_gw, lin_gb, lane, false);
  groupnorm(o1, lin_gw, lin_gb, lane, false);

#pragma unroll
  for (int r = 0; r < 8; ++r)
#pragma unroll
    for (int nt = 0; nt < 8; ++nt) {
      size_t f0 = (size_t)(row0 + r + 8 * kg) * 128 + nt * 16 + ln;
      size_t f1 = (size_t)(row0 + 16 + r + 8 * kg) * 128 + nt * 16 + ln;
      accBuf[f0] = fmaxf(o0[nt][r] + agts[f0], 0.0f);   // accBuf aliases d_out
      accBuf[f1] = fmaxf(o1[nt][r] + agts[f1], 0.0f);
    }
}

extern "C" void kernel_launch(void* const* d_in, const int* in_sizes, int n_in,
                              void* d_out, int out_size, void* d_ws, size_t ws_size,
                              hipStream_t stream) {
  const float* agts     = (const float*)d_in[0];
  const float* ctx      = (const float*)d_in[1];
  const float* agt_ctrs = (const float*)d_in[2];
  const float* ctx_ctrs = (const float*)d_in[3];
  const int*   hi       = (const int*)d_in[4];
  const int*   wi       = (const int*)d_in[5];
  const float* dist_w1  = (const float*)d_in[6];
  const float* dist_b1  = (const float*)d_in[7];
  const float* dist_w2  = (const float*)d_in[8];
  const float* dist_gw  = (const float*)d_in[9];
  const float* dist_gb  = (const float*)d_in[10];
  const float* q_w      = (const float*)d_in[11];
  const float* q_gw     = (const float*)d_in[12];
  const float* q_gb     = (const float*)d_in[13];
  const float* ctx_w1   = (const float*)d_in[14];
  const float* ctx_gw   = (const float*)d_in[15];
  const float* ctx_gb   = (const float*)d_in[16];
  const float* ctx_w2   = (const float*)d_in[17];
  const float* agt_w    = (const float*)d_in[18];
  const float* norm_w   = (const float*)d_in[19];
  const float* norm_b   = (const float*)d_in[20];
  const float* lin_w    = (const float*)d_in[21];
  const float* lin_gw   = (const float*)d_in[22];
  const float* lin_gb   = (const float*)d_in[23];

  int E    = in_sizes[4];
  int nAgt = in_sizes[0] / 128;
  float*    out   = (float*)d_out;        // doubles as the message accumulator
  _Float16* frags = (_Float16*)d_ws;      // 256 KB of fragment-layout f16 weights

  build_frags<<<(FR_TOTAL + 255) / 256, 256, 0, stream>>>(
      dist_w2, q_w, ctx_w1, ctx_w2, agt_w, lin_w, frags);

  int aTiles = (nAgt + 31) / 32;
  init_acc_kernel<<<(aTiles + 1) / 2, 64, 0, stream>>>(agts, frags, out, nAgt);

  int eTiles = (E + 31) / 32;
  edge_kernel<<<(eTiles + 1) / 2, 64, 0, stream>>>(
      agts, ctx, agt_ctrs, ctx_ctrs, hi, wi, dist_w1, dist_b1,
      dist_gw, dist_gb, q_gw, q_gb, ctx_gw, ctx_gb, frags, out, E);

  final_kernel<<<(aTiles + 1) / 2, 64, 0, stream>>>(
      agts, frags, norm_w, norm_b, lin_gw, lin_gb, out, nAgt);
}
